// ALIGNNPyG_919123001699
// MI455X (gfx1250) — matrix operand order repaired and checked by reading, not compiled, not run on hardware
//
#include <hip/hip_runtime.h>
#include <hip/hip_bf16.h>

// ---------------------------------------------------------------------------
// Problem constants (match reference)
// ---------------------------------------------------------------------------
#define NN      10000     // nodes
#define NE      120000    // edges
#define NLE     240000    // line-graph edges
#define HID     128
#define EMB     64
#define BN_EPS  1e-5f

typedef __attribute__((ext_vector_type(16))) __bf16 bf16x16;
typedef __attribute__((ext_vector_type(8)))  float  floatx8;

__device__ __forceinline__ float sigf(float x) { return 1.0f / (1.0f + expf(-x)); }
__device__ __forceinline__ float siluf(float x) { return x * sigf(x); }

// ---------------------------------------------------------------------------
// WMMA GEMM:  C[M,N] = A[M,K] @ W[K,N] + bias[N]
// A, W fp32 in memory; converted to bf16; fp32 accumulate via
// v_wmma_f32_16x16x32_bf16. One 16x16 output tile per wave, wave loops over
// all N-tiles (NT <= 8). W^T staged in LDS as bf16 (zero padded to KPAD).
// ---------------------------------------------------------------------------
template <int NT, int KPAD>
__global__ void __launch_bounds__(128)
k_gemm(const float* __restrict__ A, const float* __restrict__ W,
       const float* __restrict__ bias, float* __restrict__ C, int M, int K) {
  constexpr int N = NT * 16;
  __shared__ __bf16 sW[N * KPAD];   // transposed: sW[n*KPAD + k]

  const int tid = threadIdx.x;
  for (int idx = tid; idx < N * KPAD; idx += 128) {
    int n = idx / KPAD;
    int k = idx - n * KPAD;
    sW[idx] = (k < K) ? (__bf16)W[k * N + n] : (__bf16)0.0f;
  }
  __syncthreads();

  const int lane  = tid & 31;
  const int h     = lane >> 4;     // lane half
  const int l16   = lane & 15;
  const int mtile = blockIdx.x * 4 + (tid >> 5);

  long row  = (long)mtile * 16 + l16;          // A row owned by this lane
  long arow = (row < M) ? row : (long)(M - 1); // clamp (results masked at store)
  const float* Arow = A + arow * (long)K;

  floatx8 acc[NT] = {};

  for (int kb = 0; kb < K; kb += 32) {
    bf16x16 af;
    if (kb + 32 <= K) {
#pragma unroll
      for (int e = 0; e < 16; ++e) {
        int k = kb + 16 * h + e;   // contiguous k order (matches B)
        af[e] = (__bf16)Arow[k];
      }
    } else {
#pragma unroll
      for (int e = 0; e < 16; ++e) {
        int k = kb + 16 * h + e;
        af[e] = (k < K) ? (__bf16)Arow[k] : (__bf16)0.0f;
      }
    }
#pragma unroll
    for (int nt = 0; nt < NT; ++nt) {
      const __bf16* bp = sW + (nt * 16 + l16) * KPAD + kb + 16 * h; // 32x16 B layout
      bf16x16 bfrag;
#pragma unroll
      for (int e = 0; e < 16; ++e) bfrag[e] = bp[e];
      acc[nt] = __builtin_amdgcn_wmma_f32_16x16x32_bf16(
          false, af, false, bfrag, (short)0, acc[nt], false, false);
    }
  }

  if ((long)mtile * 16 < M) {
#pragma unroll
    for (int nt = 0; nt < NT; ++nt) {
      int col = nt * 16 + l16;
      float bv = bias[col];
#pragma unroll
      for (int r = 0; r < 8; ++r) {
        long orow = (long)mtile * 16 + r + 8 * h;     // ISA 16x16 f32 C/D layout
        if (orow < M) C[orow * (long)N + col] = acc[nt][r] + bv;
      }
    }
  }
}

static inline void launch_gemm(hipStream_t s, const float* A, const float* W,
                               const float* b, float* C, int M, int K, int N) {
  dim3 blk(128);
  dim3 grd((unsigned)((M + 63) / 64));
  if (N == 128) {
    if (K <= 64)      k_gemm<8, 64><<<grd, blk, 0, s>>>(A, W, b, C, M, K);
    else if (K <= 96) k_gemm<8, 96><<<grd, blk, 0, s>>>(A, W, b, C, M, K);
    else              k_gemm<8, 128><<<grd, blk, 0, s>>>(A, W, b, C, M, K);
  } else {  // N == 64
    if (K <= 64)      k_gemm<4, 64><<<grd, blk, 0, s>>>(A, W, b, C, M, K);
    else              k_gemm<4, 160><<<grd, blk, 0, s>>>(A, W, b, C, M, K);
  }
}

// ---------------------------------------------------------------------------
// Elementwise / reduction helper kernels
// ---------------------------------------------------------------------------
__global__ void k_fill4(float* __restrict__ p, long n4, float v) {
  long i = blockIdx.x * (long)blockDim.x + threadIdx.x;
  if (i < n4) {
    float4 f = {v, v, v, v};
    ((float4*)p)[i] = f;
  }
}

__global__ void k_gather_atoms(const int* __restrict__ types,
                               const float* __restrict__ table,
                               float* __restrict__ feat, int n, int F) {
  long i = blockIdx.x * (long)blockDim.x + threadIdx.x;
  if (i >= (long)n * F) return;
  int r = (int)(i / F), c = (int)(i % F);
  feat[i] = table[(long)types[r] * F + c];
}

__global__ void k_rbf_bond(const float* __restrict__ attr, float* __restrict__ feat, int E) {
  long t = blockIdx.x * (long)blockDim.x + threadIdx.x;
  if (t >= (long)E * 160) return;
  int e = (int)(t / 160), col = (int)(t % 160);
  float d, vmin, vmax;
  int b;
  if (col < 40) { d = attr[e * 4 + 0]; vmin = 0.0f; vmax = 8.0f; b = col; }
  else {
    int tt = (col - 40) / 40;
    b = (col - 40) % 40;
    d = attr[e * 4 + 1 + tt]; vmin = 0.0f; vmax = 1.0f;
  }
  float c = vmin + (float)b * (vmax - vmin) / 39.0f;
  float gamma = 39.0f / (vmax - vmin);
  float x = d - c;
  feat[t] = expf(-gamma * x * x);
}

__global__ void k_rbf_angle(const float* __restrict__ a, float* __restrict__ feat, int E) {
  long t = blockIdx.x * (long)blockDim.x + threadIdx.x;
  if (t >= (long)E * 40) return;
  int e = (int)(t / 40), b = (int)(t % 40);
  float c = -1.0f + (float)b * 2.0f / 39.0f;
  float gamma = 39.0f / 2.0f;
  float x = a[e] - c;
  feat[t] = expf(-gamma * x * x);
}

// Column sums / sums-of-squares. Each thread owns 4 consecutive channels,
// accumulates in registers over a grid-stride row loop (float4 loads), then
// flushes once through LDS atomics + one global atomic per channel per block.
// Requires N % 4 == 0 and blockDim.x == 128.
__global__ void __launch_bounds__(128)
k_colstats(const float* __restrict__ X, const float* __restrict__ add,
           long M, int N, float* __restrict__ sums, float* __restrict__ sumsq) {
  __shared__ float ss[128], sq[128];
  int tid = threadIdx.x;
  if (tid < N) { ss[tid] = 0.0f; sq[tid] = 0.0f; }
  __syncthreads();
  const int groups = N >> 2;            // channel groups of 4
  const int rpb    = 128 / groups;      // rows per block pass (4 for N=128, 8 for N=64)
  const int rt     = tid / groups;
  const int c0     = (tid - rt * groups) * 4;
  float s0 = 0, s1 = 0, s2 = 0, s3 = 0;
  float q0 = 0, q1 = 0, q2 = 0, q3 = 0;
  for (long r = (long)blockIdx.x * rpb + rt; r < M; r += (long)gridDim.x * rpb) {
    float4 v = *(const float4*)(X + r * (long)N + c0);
    if (add) {
      const float4 a = *(const float4*)(add + r * (long)N + c0);
      v.x += a.x; v.y += a.y; v.z += a.z; v.w += a.w;
    }
    s0 += v.x; s1 += v.y; s2 += v.z; s3 += v.w;
    q0 += v.x * v.x; q1 += v.y * v.y; q2 += v.z * v.z; q3 += v.w * v.w;
  }
  atomicAdd(&ss[c0 + 0], s0); atomicAdd(&ss[c0 + 1], s1);
  atomicAdd(&ss[c0 + 2], s2); atomicAdd(&ss[c0 + 3], s3);
  atomicAdd(&sq[c0 + 0], q0); atomicAdd(&sq[c0 + 1], q1);
  atomicAdd(&sq[c0 + 2], q2); atomicAdd(&sq[c0 + 3], q3);
  __syncthreads();
  if (tid < N) { atomicAdd(&sums[tid], ss[tid]); atomicAdd(&sumsq[tid], sq[tid]); }
}

__global__ void k_bn_finalize(const float* __restrict__ sums, const float* __restrict__ sumsq,
                              const float* __restrict__ g, const float* __restrict__ b,
                              float* __restrict__ scale, float* __restrict__ shift,
                              long M, int N) {
  int c = blockIdx.x * blockDim.x + threadIdx.x;
  if (c >= N) return;
  float inv  = 1.0f / (float)M;
  float mean = sums[c] * inv;
  float var  = sumsq[c] * inv - mean * mean;
  float sc   = g[c] * rsqrtf(var + BN_EPS);
  scale[c] = sc;
  shift[c] = b[c] - mean * sc;
}

// out = (residual?) residual + silu((X (+add)) * scale + shift), float4-wide.
__global__ void k_bn_apply4(const float* __restrict__ X, const float* __restrict__ add,
                            const float* __restrict__ scale, const float* __restrict__ shift,
                            const float* __restrict__ residual, float* __restrict__ out,
                            long M, int N) {
  long i4 = blockIdx.x * (long)blockDim.x + threadIdx.x;
  long tot4 = M * (long)N >> 2;
  if (i4 >= tot4) return;
  long i = i4 * 4;
  int c = (int)(i % N);           // multiple of 4
  float4 v = *(const float4*)(X + i);
  if (add) {
    const float4 a = *(const float4*)(add + i);
    v.x += a.x; v.y += a.y; v.z += a.z; v.w += a.w;
  }
  const float4 sc = *(const float4*)(scale + c);
  const float4 sh = *(const float4*)(shift + c);
  float4 o;
  o.x = siluf(v.x * sc.x + sh.x);
  o.y = siluf(v.y * sc.y + sh.y);
  o.z = siluf(v.z * sc.z + sh.z);
  o.w = siluf(v.w * sc.w + sh.w);
  if (residual) {
    const float4 r = *(const float4*)(residual + i);
    o.x += r.x; o.y += r.y; o.z += r.z; o.w += r.w;
  }
  *(float4*)(out + i) = o;
}

// edge_message = SG[i] + DG[j] + EG[e] (EG overwritten in place with em),
// denom[i] += sigmoid(em)
__global__ void k_edge_message(const int* __restrict__ ii, const int* __restrict__ jj,
                               const float* __restrict__ SG, const float* __restrict__ DG,
                               float* __restrict__ EMEG, float* __restrict__ DEN, int E) {
  long t = blockIdx.x * (long)blockDim.x + threadIdx.x;
  if (t >= (long)E * 32) return;                 // 128 channels / 4 per thread
  int e = (int)(t >> 5);
  int c = (int)(t & 31) * 4;
  int i = ii[e], j = jj[e];
  const float4 sg = *(const float4*)(SG + (long)i * HID + c);
  const float4 dg = *(const float4*)(DG + (long)j * HID + c);
  float4 em = *(const float4*)(EMEG + (long)e * HID + c);
  em.x += sg.x + dg.x; em.y += sg.y + dg.y;
  em.z += sg.z + dg.z; em.w += sg.w + dg.w;
  *(float4*)(EMEG + (long)e * HID + c) = em;
  float* dp = DEN + (long)i * HID + c;
  atomicAdd(dp + 0, sigf(em.x));
  atomicAdd(dp + 1, sigf(em.y));
  atomicAdd(dp + 2, sigf(em.z));
  atomicAdd(dp + 3, sigf(em.w));
}

// agg[i] += sigmoid(em) * DU[j] / denom[i]
__global__ void k_node_agg(const int* __restrict__ ii, const int* __restrict__ jj,
                           const float* __restrict__ EM, const float* __restrict__ DU,
                           const float* __restrict__ DEN, float* __restrict__ AGG, int E) {
  long t = blockIdx.x * (long)blockDim.x + threadIdx.x;
  if (t >= (long)E * 32) return;
  int e = (int)(t >> 5);
  int c = (int)(t & 31) * 4;
  int i = ii[e], j = jj[e];
  const float4 em = *(const float4*)(EM + (long)e * HID + c);
  const float4 du = *(const float4*)(DU + (long)j * HID + c);
  const float4 dn = *(const float4*)(DEN + (long)i * HID + c);
  float* ap = AGG + (long)i * HID + c;
  atomicAdd(ap + 0, sigf(em.x) * du.x / dn.x);
  atomicAdd(ap + 1, sigf(em.y) * du.y / dn.y);
  atomicAdd(ap + 2, sigf(em.z) * du.z / dn.z);
  atomicAdd(ap + 3, sigf(em.w) * du.w / dn.w);
}

// out[e] = dot(Y[e,:128], w) + b  (wave32 shuffle reduction)
__global__ void k_head(const float* __restrict__ Y, const float* __restrict__ w,
                       const float* __restrict__ b, float* __restrict__ out, int E) {
  int t = blockIdx.x * blockDim.x + threadIdx.x;
  int e = t >> 5, lane = t & 31;
  if (e >= E) return;
  float s = 0.0f;
  for (int c = lane; c < HID; c += 32) s += Y[(long)e * HID + c] * w[c];
#pragma unroll
  for (int off = 16; off > 0; off >>= 1) s += __shfl_down(s, off, 32);
  if (lane == 0) out[e] = s + b[0];
}

// ---------------------------------------------------------------------------
// Host-side orchestration helpers
// ---------------------------------------------------------------------------
static inline int nblk(long n, int b) { return (int)((n + b - 1) / b); }

static void bn_silu_run(hipStream_t s, const float* X, const float* add,
                        const float* g, const float* b, const float* residual,
                        float* out, long M, int N, float* stats) {
  float* sums  = stats;
  float* sumsq = stats + 128;
  float* scale = stats + 256;
  float* shift = stats + 384;
  k_fill4<<<1, 64, 0, s>>>(stats, 64, 0.0f);
  const int rpb = 512 / N;                       // rows per block pass
  int cb = nblk(M, rpb); if (cb > 2048) cb = 2048;
  k_colstats<<<cb, 128, 0, s>>>(X, add, M, N, sums, sumsq);
  k_bn_finalize<<<1, 128, 0, s>>>(sums, sumsq, g, b, scale, shift, M, N);
  long tot4 = M * (long)N >> 2;
  k_bn_apply4<<<nblk(tot4, 256), 256, 0, s>>>(X, add, scale, shift, residual, out, M, N);
}

static void eggc_run(hipStream_t s, void* const* d_in, int pbase,
                     const int* ii, const int* jj, int n_nodes, int n_edges,
                     float* Xin, float* Ein, float* Xout, float* Eout,
                     float* tSG, float* tDG, float* tDU, float* tSU, float* tEG,
                     float* DEN, float* AGG, float* stats) {
  auto F = [&](int k) { return (const float*)d_in[pbase + k]; };
  // param order: sg.w sg.b dg.w dg.b eg.w eg.b su.w su.b du.w du.b bne.g bne.b bnn.g bnn.b
  launch_gemm(s, Xin, F(0), F(1), tSG, n_nodes, HID, HID);
  launch_gemm(s, Xin, F(2), F(3), tDG, n_nodes, HID, HID);
  launch_gemm(s, Ein, F(4), F(5), tEG, n_edges, HID, HID);
  launch_gemm(s, Xin, F(6), F(7), tSU, n_nodes, HID, HID);
  launch_gemm(s, Xin, F(8), F(9), tDU, n_nodes, HID, HID);
  long nd4 = (long)n_nodes * HID >> 2;
  k_fill4<<<nblk(nd4, 256), 256, 0, s>>>(DEN, nd4, 1e-9f);
  k_fill4<<<nblk(nd4, 256), 256, 0, s>>>(AGG, nd4, 0.0f);
  long et = (long)n_edges * 32;
  k_edge_message<<<nblk(et, 256), 256, 0, s>>>(ii, jj, tSG, tDG, tEG, DEN, n_edges);
  k_node_agg<<<nblk(et, 256), 256, 0, s>>>(ii, jj, tEG, tDU, DEN, AGG, n_edges);
  // e_out = e + bn_silu(edge_message)
  bn_silu_run(s, tEG, nullptr, F(10), F(11), Ein, Eout, n_edges, HID, stats);
  // x_out = x + bn_silu(src_update(x) + agg)
  bn_silu_run(s, tSU, AGG, F(12), F(13), Xin, Xout, n_nodes, HID, stats);
}

// ---------------------------------------------------------------------------
// Entry point
// ---------------------------------------------------------------------------
extern "C" void kernel_launch(void* const* d_in, const int* in_sizes, int n_in,
                              void* d_out, int out_size, void* d_ws, size_t ws_size,
                              hipStream_t stream) {
  (void)in_sizes; (void)n_in; (void)out_size; (void)ws_size;

  const int*   node_types   = (const int*)d_in[0];
  const float* edge_attr    = (const float*)d_in[1];
  const float* lg_edge_attr = (const float*)d_in[2];
  const int*   ei           = (const int*)d_in[3];
  const int*   lei          = (const int*)d_in[4];
  const int* e_i = ei;        const int* e_j = ei + NE;
  const int* l_i = lei;       const int* l_j = lei + NLE;
  auto F = [&](int k) { return (const float*)d_in[k]; };

  // Workspace bump allocator (256-byte aligned blocks)
  char* base = (char*)d_ws;
  size_t off = 0;
  auto alloc = [&](long n) -> float* {
    float* p = (float*)(base + off);
    off += (size_t)n * sizeof(float);
    off = (off + 255) & ~(size_t)255;
    return p;
  };
  float* X     = alloc((long)NN * HID);
  float* Y     = alloc((long)NE * HID);
  float* Z     = alloc((long)NLE * HID);
  float* tSG   = alloc((long)NE * HID);
  float* tDG   = alloc((long)NE * HID);
  float* tDU   = alloc((long)NE * HID);
  float* tSU   = alloc((long)NE * HID);
  float* tEG   = alloc((long)NLE * HID);   // also reused as feature buffer
  float* DEN   = alloc((long)NE * HID);
  float* AGG   = alloc((long)NE * HID);
  float* stats = alloc(1024);

  // ---- atom embedding: gather -> linear(92->128) -> BN+SiLU ----
  k_gather_atoms<<<nblk((long)NN * 92, 256), 256, 0, stream>>>(node_types, F(5), tEG, NN, 92);
  launch_gemm(stream, tEG, F(6), F(7), X, NN, 92, HID);
  bn_silu_run(stream, X, nullptr, F(8), F(9), nullptr, X, NN, HID, stats);

  // ---- bond embedding: RBF(160) -> 160->64 -> BN+SiLU -> 64->128 -> BN+SiLU ----
  k_rbf_bond<<<nblk((long)NE * 160, 256), 256, 0, stream>>>(edge_attr, tEG, NE);
  launch_gemm(stream, tEG, F(10), F(11), tSG, NE, 160, EMB);
  bn_silu_run(stream, tSG, nullptr, F(12), F(13), nullptr, tSG, NE, EMB, stats);
  launch_gemm(stream, tSG, F(14), F(15), Y, NE, EMB, HID);
  bn_silu_run(stream, Y, nullptr, F(16), F(17), nullptr, Y, NE, HID, stats);

  // ---- angle embedding: RBF(40) -> 40->64 -> BN+SiLU -> 64->128 -> BN+SiLU ----
  k_rbf_angle<<<nblk((long)NLE * 40, 256), 256, 0, stream>>>(lg_edge_attr, tEG, NLE);
  launch_gemm(stream, tEG, F(18), F(19), tSG, NLE, 40, EMB);
  bn_silu_run(stream, tSG, nullptr, F(20), F(21), nullptr, tSG, NLE, EMB, stats);
  launch_gemm(stream, tSG, F(22), F(23), Z, NLE, EMB, HID);
  bn_silu_run(stream, Z, nullptr, F(24), F(25), nullptr, Z, NLE, HID, stats);

  // ---- ALIGNN layers (node EGGC on graph, edge EGGC on line graph) ----
  for (int l = 0; l < 2; ++l) {
    int nb = 26 + l * 28;       // node eggc params (14 leaves)
    int eb = nb + 14;           // edge eggc params
    // x, m = eggc(node, edge_index, x, y)   (m written into Y)
    eggc_run(stream, d_in, nb, e_i, e_j, NN, NE, X, Y, X, Y,
             tSG, tDG, tDU, tSU, tEG, DEN, AGG, stats);
    // y, z = eggc(edge, lg_edge_index, m, z)
    eggc_run(stream, d_in, eb, l_i, l_j, NE, NLE, Y, Z, Y, Z,
             tSG, tDG, tDU, tSU, tEG, DEN, AGG, stats);
  }

  // ---- GCN layers ----
  for (int l = 0; l < 2; ++l) {
    int gb = 82 + l * 14;
    eggc_run(stream, d_in, gb, e_i, e_j, NN, NE, X, Y, X, Y,
             tSG, tDG, tDU, tSU, tEG, DEN, AGG, stats);
  }

  // ---- head: [NE,128] @ [128,1] + b ----
  k_head<<<nblk((long)NE * 32, 256), 256, 0, stream>>>(Y, F(110), F(111),
                                                       (float*)d_out, NE);
}
